// RGIN_31190052504405
// MI455X (gfx1250) — matrix operand order repaired and checked by reading, not compiled, not run on hardware
//
#include <hip/hip_runtime.h>
#include <hip/hip_bf16.h>
#include <stdint.h>

#define N_NODES 50000
#define NFEAT   256
#define NHID    512
#define NCLASS  64
#define N_EDGES 800000

typedef __attribute__((ext_vector_type(16))) __bf16 v16bf;
typedef __attribute__((ext_vector_type(8)))  float  v8f;

union FragB { uint4 q[2]; v16bf v; };

__device__ __forceinline__ unsigned short f2bf(float f) {
    unsigned u = __float_as_uint(f);
    u += 0x7FFFu + ((u >> 16) & 1u);          // round-to-nearest-even
    return (unsigned short)(u >> 16);
}

// 16-byte async global->LDS copy (gfx1250, ASYNCcnt-tracked)
__device__ __forceinline__ void async_ld16(unsigned ldsOff, const void* gaddr) {
    asm volatile("global_load_async_to_lds_b128 %0, %1, off"
                 :
                 : "v"(ldsOff), "v"(gaddr)
                 : "memory");
}
__device__ __forceinline__ void wait_async0() {
    asm volatile("s_wait_asynccnt 0x0" ::: "memory");
}

// ---------------------------------------------------------------------------
// Weight prep: Wt[n*K + k] = bf16(W[k*Nout + n])   (transpose + convert)
// ---------------------------------------------------------------------------
__global__ void wprep_kernel(const float* __restrict__ W,
                             unsigned short* __restrict__ Wt,
                             int K, int Nout) {
    int idx = blockIdx.x * blockDim.x + threadIdx.x;
    if (idx >= K * Nout) return;
    int n = idx / K, k = idx - n * K;
    Wt[idx] = f2bf(W[(size_t)k * Nout + n]);
}

// ---------------------------------------------------------------------------
// float4 copy (dst = src), n4 = element count / 4
// ---------------------------------------------------------------------------
__global__ void copy4_kernel(float* __restrict__ dst, const float* __restrict__ src,
                             long long n4) {
    long long i = (long long)blockIdx.x * blockDim.x + threadIdx.x;
    if (i >= n4) return;
    ((float4*)dst)[i] = ((const float4*)src)[i];
}

// ---------------------------------------------------------------------------
// f32 -> bf16 convert (vectorized x4)
// ---------------------------------------------------------------------------
__global__ void cvt_bf16_kernel(unsigned short* __restrict__ dst,
                                const float* __restrict__ src, long long n4) {
    long long i = (long long)blockIdx.x * blockDim.x + threadIdx.x;
    if (i >= n4) return;
    float4 v = ((const float4*)src)[i];
    ushort4 o;
    o.x = f2bf(v.x); o.y = f2bf(v.y); o.z = f2bf(v.z); o.w = f2bf(v.w);
    ((ushort4*)dst)[i] = o;
}

// ---------------------------------------------------------------------------
// Edge scatter: out[dst[e]] += feat[src[e]]  (one thread = one edge x 4 feats)
// ---------------------------------------------------------------------------
__global__ void scatter_add_kernel(const float* __restrict__ feat,
                                   float* __restrict__ out,
                                   const int* __restrict__ src,
                                   const int* __restrict__ dst,
                                   int nEdges, int D, int chunkShift) {
    long long tid = (long long)blockIdx.x * blockDim.x + threadIdx.x;
    long long total = (long long)nEdges << chunkShift;
    if (tid >= total) return;
    int e = (int)(tid >> chunkShift);
    int c = (int)(tid & ((1 << chunkShift) - 1));
    int s = src[e], d = dst[e];
    float4 v = *(const float4*)(feat + (size_t)s * D + (c << 2));
    float* p = out + (size_t)d * D + (c << 2);
    atomicAdd(p + 0, v.x); atomicAdd(p + 1, v.y);
    atomicAdd(p + 2, v.z); atomicAdd(p + 3, v.w);
}

// ---------------------------------------------------------------------------
// WMMA GEMM with LDS-staged B tile.
//   C[nRows x nOut] = A[nRows x K](bf16) @ Wt^T   (Wt is [nOut x K] bf16)
// Grid: (ceil(Mtiles/8), nOut/64). Block = 8 waves; all waves share one
// 64-column B tile staged into LDS via global_load_async_to_lds_b128;
// each wave owns one 16-row M tile (4 accumulators = 16x64 output).
// Per k-step: load A frag (2x global b128) + ALL FOUR B frags (8x ds b128),
// then issue 4 back-to-back v_wmma so DS/VMEM latency overlaps the XDL pipe.
// ---------------------------------------------------------------------------
template<int K, int RELU, int OUTBF, int DUAL>
__global__ __launch_bounds__(256) void gemm_wmma_kernel(
    const unsigned short* __restrict__ A,
    const unsigned short* __restrict__ Wt,
    const float* __restrict__ bias,
    float* __restrict__ C0, float* __restrict__ C1,
    unsigned short* __restrict__ Cbf,
    int nRows, int nOut) {
    __shared__ unsigned short sB[64 * K];             // 64 output cols x K (bf16)

    const int lane = threadIdx.x & 31;
    const int wave = threadIdx.x >> 5;
    const int nTile = blockIdx.y;

    // ---- cooperative async stage of the 64xK B tile into LDS ----
    {
        const unsigned short* src = Wt + (size_t)(nTile << 6) * K;
        const unsigned ldsBase = (unsigned)(unsigned long long)&sB[0];
        constexpr int chunksPerThread = (64 * K / 8) / 256;   // 16B chunks
#pragma unroll
        for (int i = 0; i < chunksPerThread; ++i) {
            const int chunk = i * 256 + threadIdx.x;
            async_ld16(ldsBase + chunk * 16, src + chunk * 8);
        }
        wait_async0();
        __syncthreads();
    }

    const int mTile = blockIdx.x * 8 + wave;
    if (mTile >= (nRows >> 4)) return;                // wave-uniform, after barrier

    const int half = lane >> 4;
    const int r16  = lane & 15;

    const unsigned short* aPtr = A + (size_t)((mTile << 4) + r16) * K + half * 8;
    const unsigned short* bLds[4];
#pragma unroll
    for (int nt = 0; nt < 4; ++nt)
        bLds[nt] = sB + (nt * 16 + r16) * K + half * 8;

    v8f acc[4] = {};

#pragma unroll
    for (int k = 0; k < K; k += 32) {
        FragB a;
        a.q[0] = *(const uint4*)(aPtr + k);
        a.q[1] = *(const uint4*)(aPtr + k + 16);
        FragB b[4];
#pragma unroll
        for (int nt = 0; nt < 4; ++nt) {
            b[nt].q[0] = *(const uint4*)(bLds[nt] + k);     // ds_load_b128
            b[nt].q[1] = *(const uint4*)(bLds[nt] + k + 16);
        }
#pragma unroll
        for (int nt = 0; nt < 4; ++nt)
            acc[nt] = __builtin_amdgcn_wmma_f32_16x16x32_bf16(
                false, a.v, false, b[nt].v, (short)0, acc[nt], false, false);
    }

    // C/D layout: VGPR r, lane<16: (M=r, N=lane); lane>=16: (M=8+r, N=lane-16)
    const int rowBase = (mTile << 4) + half * 8;
    const int colBase = (nTile << 6) + r16;
#pragma unroll
    for (int nt = 0; nt < 4; ++nt) {
        const int c = colBase + nt * 16;
        const float bv = bias[c];
#pragma unroll
        for (int r = 0; r < 8; ++r) {
            float v = acc[nt][r] + bv;
            if (RELU) v = fmaxf(v, 0.0f);
            const size_t idx = (size_t)(rowBase + r) * nOut + c;
            if (OUTBF) {
                Cbf[idx] = f2bf(v);
            } else {
                C0[idx] = v;
                if (DUAL) C1[idx] = v;
            }
        }
    }
}

// ---------------------------------------------------------------------------
// In-place log_softmax over rows of 64 (one wave32 per row, 2 cols/lane)
// ---------------------------------------------------------------------------
__global__ void logsoftmax64_kernel(float* __restrict__ out, int nRows) {
    int lane = threadIdx.x & 31;
    int row = blockIdx.x * (blockDim.x >> 5) + (threadIdx.x >> 5);
    if (row >= nRows) return;
    float* p = out + (size_t)row * 64;
    float a = p[lane], b = p[lane + 32];
    float m = fmaxf(a, b);
#pragma unroll
    for (int o = 16; o > 0; o >>= 1) m = fmaxf(m, __shfl_xor(m, o, 32));
    float s = __expf(a - m) + __expf(b - m);
#pragma unroll
    for (int o = 16; o > 0; o >>= 1) s += __shfl_xor(s, o, 32);
    float l = m + __logf(s);
    p[lane] = a - l;
    p[lane + 32] = b - l;
}

// ---------------------------------------------------------------------------
extern "C" void kernel_launch(void* const* d_in, const int* in_sizes, int n_in,
                              void* d_out, int out_size, void* d_ws, size_t ws_size,
                              hipStream_t stream) {
    const float* x   = (const float*)d_in[0];
    const int*   ei  = (const int*)d_in[1];
    const int*   srcIdx = ei;             // edge_index[0]
    const int*   dstIdx = ei + N_EDGES;   // edge_index[1]
    const float* W1a = (const float*)d_in[2];
    const float* b1a = (const float*)d_in[3];
    const float* W2a = (const float*)d_in[4];
    const float* b2a = (const float*)d_in[5];
    const float* W1b = (const float*)d_in[6];
    const float* b1b = (const float*)d_in[7];
    const float* W2b = (const float*)d_in[8];
    const float* b2b = (const float*)d_in[9];

    // ---- workspace carve (regions overlapped by liveness) ----
    uint8_t* ws = (uint8_t*)d_ws;
    size_t off = 0;
    auto take = [&](size_t bytes) -> void* {
        void* p = ws + off;
        off += (bytes + 255) & ~(size_t)255;
        return p;
    };
    // region0: hin1 (N x 256 f32) -> later h1 (N x 512 f32)
    float* region0 = (float*)take((size_t)N_NODES * NHID * 4);
    float* hin1 = region0;
    float* h1   = region0;
    // region1: hin1bf (N x 256 bf16) -> later hin2 (N x 512 f32)
    void* region1 = take((size_t)N_NODES * NHID * 4);
    unsigned short* hin1bf = (unsigned short*)region1;
    float*          hin2   = (float*)region1;
    // region2: t1bf (N x 512 bf16) -> later hin2bf (N x 512 bf16)
    void* region2 = take((size_t)N_NODES * NHID * 2);
    unsigned short* t1bf   = (unsigned short*)region2;
    unsigned short* hin2bf = (unsigned short*)region2;
    // region3: t2bf (N x 64 bf16)
    unsigned short* t2bf = (unsigned short*)take((size_t)N_NODES * NCLASS * 2);
    // transposed bf16 weights
    unsigned short* W1aT = (unsigned short*)take((size_t)NFEAT * NHID * 2);
    unsigned short* W2aT = (unsigned short*)take((size_t)NHID * NHID * 2);
    unsigned short* W1bT = (unsigned short*)take((size_t)NHID * NCLASS * 2);
    unsigned short* W2bT = (unsigned short*)take((size_t)NCLASS * NCLASS * 2);

    const int TPB = 256;
    const int M_TILES = N_NODES / 16;                 // 3125 (exact)
    const unsigned M_BLKS = (M_TILES + 7) / 8;        // 391

    // 1) weight prep
    wprep_kernel<<<(NFEAT * NHID + TPB - 1) / TPB, TPB, 0, stream>>>(W1a, W1aT, NFEAT, NHID);
    wprep_kernel<<<(NHID * NHID + TPB - 1) / TPB, TPB, 0, stream>>>(W2a, W2aT, NHID, NHID);
    wprep_kernel<<<(NHID * NCLASS + TPB - 1) / TPB, TPB, 0, stream>>>(W1b, W1bT, NHID, NCLASS);
    wprep_kernel<<<(NCLASS * NCLASS + TPB - 1) / TPB, TPB, 0, stream>>>(W2b, W2bT, NCLASS, NCLASS);

    // 2) hin1 = x
    {
        long long n4 = (long long)N_NODES * NFEAT / 4;
        copy4_kernel<<<(unsigned)((n4 + TPB - 1) / TPB), TPB, 0, stream>>>(hin1, x, n4);
    }
    // 3) hin1 += scatter(x)
    {
        long long total = (long long)N_EDGES << 6;    // 64 chunks of 4 (D=256)
        scatter_add_kernel<<<(unsigned)((total + TPB - 1) / TPB), TPB, 0, stream>>>(
            x, hin1, srcIdx, dstIdx, N_EDGES, NFEAT, 6);
    }
    // 4) hin1bf = bf16(hin1)
    {
        long long n4 = (long long)N_NODES * NFEAT / 4;
        cvt_bf16_kernel<<<(unsigned)((n4 + TPB - 1) / TPB), TPB, 0, stream>>>(hin1bf, hin1, n4);
    }
    // 5) t1bf = relu(hin1bf @ W1a + b1a)             [N x 512, bf16 out]
    gemm_wmma_kernel<NFEAT, 1, 1, 0><<<dim3(M_BLKS, NHID / 64), TPB, 0, stream>>>(
        hin1bf, W1aT, b1a, nullptr, nullptr, t1bf, N_NODES, NHID);
    // 6) h1 = hin2 = t1bf @ W2a + b2a                [N x 512, f32 dual-store]
    gemm_wmma_kernel<NHID, 0, 0, 1><<<dim3(M_BLKS, NHID / 64), TPB, 0, stream>>>(
        t1bf, W2aT, b2a, h1, hin2, nullptr, N_NODES, NHID);
    // 7) hin2 += scatter(h1)
    {
        long long total = (long long)N_EDGES << 7;    // 128 chunks of 4 (D=512)
        scatter_add_kernel<<<(unsigned)((total + TPB - 1) / TPB), TPB, 0, stream>>>(
            h1, hin2, srcIdx, dstIdx, N_EDGES, NHID, 7);
    }
    // 8) hin2bf = bf16(hin2)
    {
        long long n4 = (long long)N_NODES * NHID / 4;
        cvt_bf16_kernel<<<(unsigned)((n4 + TPB - 1) / TPB), TPB, 0, stream>>>(hin2bf, hin2, n4);
    }
    // 9) t2bf = relu(hin2bf @ W1b + b1b)             [N x 64, bf16 out]
    gemm_wmma_kernel<NHID, 1, 1, 0><<<dim3(M_BLKS, 1), TPB, 0, stream>>>(
        hin2bf, W1bT, b1b, nullptr, nullptr, t2bf, N_NODES, NCLASS);
    // 10) d_out = t2bf @ W2b + b2b                   [N x 64, f32]
    gemm_wmma_kernel<NCLASS, 0, 0, 0><<<dim3(M_BLKS, 1), TPB, 0, stream>>>(
        t2bf, W2bT, b2b, (float*)d_out, nullptr, nullptr, N_NODES, NCLASS);
    // 11) in-place log_softmax over rows of 64
    logsoftmax64_kernel<<<(N_NODES + 7) / 8, TPB, 0, stream>>>((float*)d_out, N_NODES);
}
